// L7_77206332113747
// MI455X (gfx1250) — matrix operand order repaired and checked by reading, not compiled, not run on hardware
//
#include <hip/hip_runtime.h>

// ---------------------------------------------------------------------------
// CDNA5 (gfx1250) WMMA types
// ---------------------------------------------------------------------------
typedef __attribute__((ext_vector_type(16))) _Float16 v16h;
typedef __attribute__((ext_vector_type(8)))  _Float16 v8h;
typedef __attribute__((ext_vector_type(8)))  float    v8f;

#define GEMM_WAVES 4
#define GEMM_BLOCK (GEMM_WAVES * 32)

// Compile-time activation: 0 = identity (no bias), 1 = ELU, 2 = sigmoid
template <int ACT>
__device__ __forceinline__ float act_apply_t(float x) {
  if constexpr (ACT == 1) {
    const float e = __expf(x) - 1.f;
    return x > 0.f ? x : e;                         // v_exp + v_cndmask
  } else if constexpr (ACT == 2) {
    return __builtin_amdgcn_rcpf(1.f + __expf(-x)); // v_exp + v_rcp
  }
  return x;
}

// ---------------------------------------------------------------------------
// Build x0 = concat(one_hot[N,20], features[N,44]) -> [N,64] f32
// ---------------------------------------------------------------------------
__global__ void concat_kernel(const float* __restrict__ oh, const float* __restrict__ ft,
                              float* __restrict__ x, int N) {
  int t = blockIdx.x * blockDim.x + threadIdx.x;
  int n = t >> 6, c = t & 63;
  if (n >= N) return;
  float v = (c < 20) ? oh[n * 20 + c] : ft[n * 44 + (c - 20)];
  x[(size_t)n * 64 + c] = v;
}

// ---------------------------------------------------------------------------
// Pad + TRANSPOSE weight [K,M] f32 -> Wt[Mp,Kp] f16 (W^T row-major, zero fill),
// bias [M] -> [Mp] f32. Transposed layout makes each lane's 16 B-halves
// two contiguous runs of 8 -> two b128 loads per WMMA instead of 16 strided u16 loads.
// ---------------------------------------------------------------------------
__global__ void pad_weight_kernel(const float* __restrict__ W, const float* __restrict__ b,
                                  _Float16* __restrict__ Wt, float* __restrict__ bp,
                                  int K, int M, int Kp, int Mp) {
  int t = blockIdx.x * blockDim.x + threadIdx.x;
  if (t < Kp * Mp) {
    int m = t / Kp, k = t % Kp;            // Wt row-major [Mp, Kp]
    float v = (k < K && m < M) ? W[k * M + m] : 0.f;
    Wt[t] = (_Float16)v;
  }
  if (t < Mp) bp[t] = (t < M) ? b[t] : 0.f;
}

__global__ void zero_kernel(float* __restrict__ p, size_t n) {
  size_t t = (size_t)blockIdx.x * blockDim.x + threadIdx.x;
  if (t < n) p[t] = 0.f;
}

// ---------------------------------------------------------------------------
// Y = act( X[N,KP](f32, ld=KP) @ W[KP,MP](f16, stored as Wt[MP,KP]) [+ bias] ).
// Fully static shapes: KP, MP, ACT, FINAL are template parameters, so
//  - all A fragments (KP/32 x v16h) are register-resident, loaded once
//  - column-tile count is constexpr; col<MP store guard provably true
//  - row stride LDY is constexpr -> 8 stores use immediate offsets off one base
//  - ACT==0 (graph-conv support GEMM) skips the bias load+add entirely
//    (reference adds the gc bias AFTER the SpMM, not here)
// A layout (ISA 7.12.2, 16-bit A 16x32): lane&15 = row; lanes 0-15 hold
// K-offsets {0..7,16..23}, lanes 16-31 hold {8..15,24..31}.
// B layout: lane&15 = col; lanes 0-15 K={0..7,16..23}, lanes 16-31 K={8..15,24..31}.
// C/D: VGPR r, lane l -> row = r + 8*(l>>4), col = l&15.
// ---------------------------------------------------------------------------
template <int KP, int MP, int ACT, bool FINAL>
__global__ void __launch_bounds__(GEMM_BLOCK)
gemm_bias_act_wmma(const float* __restrict__ X, const _Float16* __restrict__ Wt,
                   const float* __restrict__ bias, float* __restrict__ Y, int N) {
  constexpr int KF  = KP / 32;           // # of 16x32 A fragments
  constexpr int LDY = FINAL ? 1 : MP;    // output row stride
  constexpr int COLTILES = MP / 16;
  const int lane = threadIdx.x & 31;
  const int wave = threadIdx.x >> 5;
  const int row0 = blockIdx.x * 16;
  const int arow = row0 + (lane & 15);
  const bool avalid = arow < N;
  const bool fullTile = (row0 + 16) <= N;          // block-uniform
  const float* __restrict__ xrow = X + (size_t)arow * KP;
  const int koff = (lane >> 4) * 8;      // A sub-K offset for upper half-wave
  const int bk0  = (lane >> 4) * 8;      // B K offset for upper half-wave
  const int bcol = lane & 15;

  // ---- load all A fragments once (register-resident across column tiles) ----
  v16h afrag[KF];
#pragma unroll
  for (int f = 0; f < KF; ++f) {
    if (avalid) {
      const int kb = f * 32;
      const float4 x0 = *reinterpret_cast<const float4*>(xrow + kb + koff);
      const float4 x1 = *reinterpret_cast<const float4*>(xrow + kb + koff + 4);
      const float4 x2 = *reinterpret_cast<const float4*>(xrow + kb + 16 + koff);
      const float4 x3 = *reinterpret_cast<const float4*>(xrow + kb + 16 + koff + 4);
      v16h a;
      a[0]=(_Float16)x0.x;  a[1]=(_Float16)x0.y;  a[2]=(_Float16)x0.z;  a[3]=(_Float16)x0.w;
      a[4]=(_Float16)x1.x;  a[5]=(_Float16)x1.y;  a[6]=(_Float16)x1.z;  a[7]=(_Float16)x1.w;
      a[8]=(_Float16)x2.x;  a[9]=(_Float16)x2.y;  a[10]=(_Float16)x2.z; a[11]=(_Float16)x2.w;
      a[12]=(_Float16)x3.x; a[13]=(_Float16)x3.y; a[14]=(_Float16)x3.z; a[15]=(_Float16)x3.w;
      afrag[f] = a;
    } else {
      v16h a;
#pragma unroll
      for (int i = 0; i < 16; ++i) a[i] = (_Float16)0.f;
      afrag[f] = a;
    }
  }

  // ---- column tiles ----
  for (int ct = wave; ct < COLTILES; ct += GEMM_WAVES) {
    v8f acc = {};
    const int col = ct * 16 + bcol;
    const _Float16* __restrict__ wrow = Wt + (size_t)col * KP + bk0;  // contiguous K
#pragma unroll
    for (int f = 0; f < KF; ++f) {
      const v8h b0 = *reinterpret_cast<const v8h*>(wrow + f * 32);
      const v8h b1 = *reinterpret_cast<const v8h*>(wrow + f * 32 + 16);
      const v16h b = __builtin_shufflevector(b0, b1, 0,1,2,3,4,5,6,7,8,9,10,11,12,13,14,15);
      // (neg_a, A, neg_b, B, c_mod, C, reuse_a, reuse_b)
      acc = __builtin_amdgcn_wmma_f32_16x16x32_f16(false, afrag[f], false, b,
                                                   (short)0, acc, false, false);
    }
    float bv = 0.f;
    if constexpr (ACT != 0) bv = bias[col];
    const int rbase = row0 + ((lane >> 4) << 3);
    float* __restrict__ yp = Y + (size_t)rbase * LDY + col;
    if (fullTile) {
      bool doStore = true;
      if constexpr (FINAL) doStore = (col < 1);
      if (doStore) {     // single predicate; stores are base + immediate offsets
#pragma unroll
        for (int r = 0; r < 8; ++r)
          yp[r * LDY] = act_apply_t<ACT>(acc[r] + bv);
      }
    } else {
#pragma unroll
      for (int r = 0; r < 8; ++r) {
        bool g = (rbase + r) < N;
        if constexpr (FINAL) g = g && (col < 1);
        if (g) yp[r * LDY] = act_apply_t<ACT>(acc[r] + bv);
      }
    }
  }
}

template <int KP, int MP, int ACT, bool FINAL = false>
static inline void run_gemm(hipStream_t s, int grid, const float* X, const _Float16* Wt,
                            const float* b, float* Y, int N) {
  gemm_bias_act_wmma<KP, MP, ACT, FINAL><<<grid, GEMM_BLOCK, 0, s>>>(X, Wt, b, Y, N);
}

// ---------------------------------------------------------------------------
// SpMM scatter: y[rows[e], :] += vals[e] * x[cols[e], :]
// thread = (edge, 4-feature chunk); consecutive threads -> coalesced gather.
// Rows are L2-resident (<=102MB activations vs 192MB L2).
// ---------------------------------------------------------------------------
__global__ void spmm_kernel(const int* __restrict__ rows, const int* __restrict__ cols,
                            const float* __restrict__ vals, const float* __restrict__ x,
                            float* __restrict__ y, int E, int chunks, int ld) {
  int t = blockIdx.x * blockDim.x + threadIdx.x;
  if (t >= E * chunks) return;
  int e  = t / chunks;
  int f4 = (t - e * chunks) * 4;
  int r = rows[e], c = cols[e];
  float v = vals[e];
  const float4 xv = *reinterpret_cast<const float4*>(x + (size_t)c * ld + f4);
  float* yp = y + (size_t)r * ld + f4;
  atomicAdd(yp + 0, v * xv.x);
  atomicAdd(yp + 1, v * xv.y);
  atomicAdd(yp + 2, v * xv.z);
  atomicAdd(yp + 3, v * xv.w);
}

// Bias + ELU on the SpMM accumulator (x = ELU(A@support + b))
__global__ void bias_elu_kernel(float* __restrict__ y, const float* __restrict__ b,
                                size_t total, int Mp) {
  size_t t = (size_t)blockIdx.x * blockDim.x + threadIdx.x;
  if (t >= total) return;
  int m = (int)(t % Mp);
  y[t] = act_apply_t<1>(y[t] + b[m]);
}

// ---------------------------------------------------------------------------
// Host orchestration
// ---------------------------------------------------------------------------
extern "C" void kernel_launch(void* const* d_in, const int* in_sizes, int n_in,
                              void* d_out, int out_size, void* d_ws, size_t ws_size,
                              hipStream_t stream) {
  const float* one_hot  = (const float*)d_in[0];
  const float* features = (const float*)d_in[1];
  // d_in[2] = gemme_features (unused, as in the reference)
  const int*   a_rows   = (const int*)d_in[3];
  const int*   a_cols   = (const int*)d_in[4];
  const float* a_vals   = (const float*)d_in[5];

  const int N = in_sizes[0] / 20;
  const int E = in_sizes[3];

  // 17 (W,b) pairs in params-dict insertion order: enc1..4, gc0..7, fin1..5
  const float* Wraw[17]; const float* braw[17];
  {
    int idx = 6;
    for (int i = 0; i < 17; ++i) { Wraw[i] = (const float*)d_in[idx++]; braw[i] = (const float*)d_in[idx++]; }
  }
  static const int DIN[17]  = {64,32,64,128,  256,200,150,128,100,64,32,16,  8,16,32,16,8};
  static const int DOUT[17] = {32,64,128,256, 200,150,128,100,64,32,16,8,   16,32,16,8,1};
  int Kp[17], Mp[17];
  for (int i = 0; i < 17; ++i) { Kp[i] = (DIN[i] + 31) & ~31; Mp[i] = (DOUT[i] + 31) & ~31; }

  // Workspace carve: 3 activation buffers [N,256] f32 + padded transposed weights
  char* ws = (char*)d_ws;
  size_t off = 0;
  auto carve = [&](size_t bytes) { void* p = ws + off; off += (bytes + 255) & ~(size_t)255; return p; };
  const size_t actBytes = (size_t)N * 256 * sizeof(float);
  float* act0 = (float*)carve(actBytes);
  float* act1 = (float*)carve(actBytes);
  float* act2 = (float*)carve(actBytes);
  _Float16* Wp[17]; float* bp[17];
  for (int i = 0; i < 17; ++i) {
    Wp[i] = (_Float16*)carve((size_t)Kp[i] * Mp[i] * sizeof(_Float16));
    bp[i] = (float*)carve((size_t)Mp[i] * sizeof(float));
  }

  // Pad/transpose/convert all weights (tiny; deterministic every call)
  for (int i = 0; i < 17; ++i) {
    int tot = Kp[i] * Mp[i];
    pad_weight_kernel<<<(tot + 255) / 256, 256, 0, stream>>>(
        Wraw[i], braw[i], Wp[i], bp[i], DIN[i], DOUT[i], Kp[i], Mp[i]);
  }

  // x0 = concat
  concat_kernel<<<(int)(((size_t)N * 64 + 255) / 256), 256, 0, stream>>>(one_hot, features, act0, N);

  const int grid = (N + 15) / 16;
  float* cur = act0;
  float* nxt = act1;
  float* accb = act2;
  auto swap2 = [&]() { float* t = cur; cur = nxt; nxt = t; };

  // Dense encoder: 4x GEMM + bias + ELU
  run_gemm<64, 32, 1>(stream, grid, cur, Wp[0], bp[0], nxt, N); swap2();
  run_gemm<32, 64, 1>(stream, grid, cur, Wp[1], bp[1], nxt, N); swap2();
  run_gemm<64, 128, 1>(stream, grid, cur, Wp[2], bp[2], nxt, N); swap2();
  run_gemm<128, 256, 1>(stream, grid, cur, Wp[3], bp[3], nxt, N); swap2();
  // cur = encoder output (ld 256); nxt = scratch; accb = third buffer

  // 8 graph-conv layers: support = x@W (no bias) ; acc = A@support ; x = ELU(acc+b)
  auto gc_rest = [&](int li, int MpV) {
    const size_t tot = (size_t)N * MpV;
    zero_kernel<<<(int)((tot + 255) / 256), 256, 0, stream>>>(accb, tot);
    const int chunks = MpV >> 2;
    const int sThreads = E * chunks;
    spmm_kernel<<<(sThreads + 255) / 256, 256, 0, stream>>>(
        a_rows, a_cols, a_vals, nxt, accb, E, chunks, MpV);
    bias_elu_kernel<<<(int)((tot + 255) / 256), 256, 0, stream>>>(accb, bp[li], tot, MpV);
    float* old = cur; cur = accb; accb = old;   // nxt stays as GEMM scratch
  };
  run_gemm<256, 224, 0>(stream, grid, cur, Wp[4],  nullptr, nxt, N); gc_rest(4, 224);
  run_gemm<224, 160, 0>(stream, grid, cur, Wp[5],  nullptr, nxt, N); gc_rest(5, 160);
  run_gemm<160, 128, 0>(stream, grid, cur, Wp[6],  nullptr, nxt, N); gc_rest(6, 128);
  run_gemm<128, 128, 0>(stream, grid, cur, Wp[7],  nullptr, nxt, N); gc_rest(7, 128);
  run_gemm<128, 64, 0>(stream, grid, cur, Wp[8],  nullptr, nxt, N); gc_rest(8, 64);
  run_gemm<64, 32, 0>(stream, grid, cur, Wp[9],  nullptr, nxt, N); gc_rest(9, 32);
  run_gemm<32, 32, 0>(stream, grid, cur, Wp[10], nullptr, nxt, N); gc_rest(10, 32);
  run_gemm<32, 32, 0>(stream, grid, cur, Wp[11], nullptr, nxt, N); gc_rest(11, 32);

  // Final head: 4x GEMM+ELU, then GEMM+sigmoid -> d_out [N,1]
  run_gemm<32, 32, 1>(stream, grid, cur, Wp[12], bp[12], nxt, N); swap2();
  run_gemm<32, 32, 1>(stream, grid, cur, Wp[13], bp[13], nxt, N); swap2();
  run_gemm<32, 32, 1>(stream, grid, cur, Wp[14], bp[14], nxt, N); swap2();
  run_gemm<32, 32, 1>(stream, grid, cur, Wp[15], bp[15], nxt, N); swap2();
  run_gemm<32, 32, 2, true>(stream, grid, cur, Wp[16], bp[16], (float*)d_out, N);
}
